// SinkhornLoss_20375324852626
// MI455X (gfx1250) — compile-verified, matrix operand order checked by reading, not run on hardware
//
#include <hip/hip_runtime.h>
#include <hip/hip_bf16.h>

// ---------------------------------------------------------------------------
// Debiased Sinkhorn divergence, fully fused softmin passes on gfx1250.
//   softmin(eps, C, h)[n] = -eps * log sum_m exp(h[m] - C[n,m]/eps)
//   C[n,m] = 0.5(|x_n|^2 + |y_m|^2) - <x_n,y_m>
// => out[n] = 0.5|x_n|^2 - eps * LSE_m( h'[m] + <x_n,y_m>/eps ),
//    h'[m]  = h[m] - 0.5|y_m|^2/eps.
// The <x_n,y_m> Gram tile is recomputed per pass with v_wmma_f32_16x16x32_f16
// (X,Y stay L2-resident: 8 MiB total vs 192 MiB L2), avoiding 5.6 GB of
// HBM traffic that materialized cost matrices would need.
// ---------------------------------------------------------------------------

typedef __attribute__((ext_vector_type(16))) _Float16 v16h;
typedef __attribute__((ext_vector_type(8)))  float    v8f;

union HFrag { v16h v; unsigned u[8]; };

#define TILE_ROWS 128     // rows (n) per workgroup: 8 waves x 16 rows
#define CHUNK_COLS 128    // columns (m) staged per LDS chunk
#define DIM 64            // feature dim (hardcoded from reference)
#define LDS_STRIDE 66     // halfs per row (padding breaks 64-bank conflicts)

__device__ __forceinline__ float fexp2(float x) {
  return __builtin_amdgcn_exp2f(x);   // v_exp_f32 (hardware exp2)
}

__device__ __forceinline__ unsigned packh2(float x, float y) {
  _Float16 hx = (_Float16)x, hy = (_Float16)y;
  unsigned short bx = __builtin_bit_cast(unsigned short, hx);
  unsigned short by = __builtin_bit_cast(unsigned short, hy);
  return ((unsigned)by << 16) | (unsigned)bx;
}

// ---------------------------------------------------------------------------
// Fused softmin pass.
//   X: [B, Nr, 64] rows; Y: [B, Mc, 64] columns.
//   h[m] = logwY[m] + potScale * potY[m]   (potScale = 1/eps or 0 for init)
//   mode 0: out[n] = softmin          mode 1: out[n] = 0.5*(prev[n] + softmin)
// ---------------------------------------------------------------------------
__global__ void __launch_bounds__(256)
sinkhorn_softmin(const float* __restrict__ X, const float* __restrict__ Y,
                 const float* __restrict__ logwY, const float* __restrict__ potY,
                 float potScale, const float* __restrict__ prevPot,
                 float* __restrict__ outPot, float eps, int Nr, int Mc, int mode)
{
  const float LOG2E = 1.44269504088896341f;
  const float LN2   = 0.69314718055994531f;

  __shared__ _Float16 sX[TILE_ROWS * LDS_STRIDE];
  __shared__ _Float16 sY[CHUNK_COLS * LDS_STRIDE];
  __shared__ float    sX2[TILE_ROWS];
  __shared__ float    sHH[CHUNK_COLS];

  const int b       = blockIdx.y;
  const int rowBase = blockIdx.x * TILE_ROWS;
  const int tid     = threadIdx.x;
  const float invEps = 1.0f / eps;

  // ---- stage X tile: f32 -> f16 into LDS, plus row |x|^2 -------------------
  {
    const int row = tid >> 1;
    const int c0  = (tid & 1) * 32;
    const float* p = X + (((size_t)b * Nr + rowBase + row) * DIM + c0);
    unsigned* d32 = (unsigned*)(sX + row * LDS_STRIDE + c0);
    float sq = 0.f;
#pragma unroll
    for (int i = 0; i < 32; i += 4) {
      float4 v = *(const float4*)(p + i);
      sq = fmaf(v.x, v.x, sq); sq = fmaf(v.y, v.y, sq);
      sq = fmaf(v.z, v.z, sq); sq = fmaf(v.w, v.w, sq);
      d32[(i >> 1) + 0] = packh2(v.x, v.y);
      d32[(i >> 1) + 1] = packh2(v.z, v.w);
    }
    sq += __shfl_xor(sq, 1, 32);
    if (!(tid & 1)) sX2[row] = sq;
  }
  __syncthreads();

  const int wave  = tid >> 5;
  const int lane  = tid & 31;
  const int halfq = lane >> 4;   // which 16-lane half
  const int l15   = lane & 15;

  // ---- A fragments (16x32 f16 layout, K-striped) held for the whole sweep --
  HFrag a0, a1;
  {
    const unsigned* pX = (const unsigned*)sX;
    const int arow = wave * 16 + l15;
#pragma unroll
    for (int v = 0; v < 8; ++v) {
      const int kb = (v < 4 ? 2 * v : 2 * v + 8) + 8 * halfq;
      a0.u[v] = pX[arow * (LDS_STRIDE / 2) + (kb >> 1)];
      a1.u[v] = pX[arow * (LDS_STRIDE / 2) + ((kb + 32) >> 1)];
    }
  }

  // online logsumexp accumulators (log2 domain); slot (lane,r) owns
  // output row r + 8*halfq of this wave's 16-row strip, columns == l15 mod 16
  float runM[8], runS[8];
#pragma unroll
  for (int r = 0; r < 8; ++r) { runM[r] = -1e30f; runS[r] = 0.f; }

  const float ie2 = invEps * LOG2E;
  const int nChunks = Mc / CHUNK_COLS;

  for (int ch = 0; ch < nChunks; ++ch) {
    const int mBase = ch * CHUNK_COLS;
    // ---- stage Y chunk + per-column h' (already x LOG2E) -------------------
    {
      const int row = tid >> 1;
      const int c0  = (tid & 1) * 32;
      const int m   = mBase + row;
      const float* p = Y + (((size_t)b * Mc + m) * DIM + c0);
      unsigned* d32 = (unsigned*)(sY + row * LDS_STRIDE + c0);
      float sq = 0.f;
#pragma unroll
      for (int i = 0; i < 32; i += 4) {
        float4 v = *(const float4*)(p + i);
        sq = fmaf(v.x, v.x, sq); sq = fmaf(v.y, v.y, sq);
        sq = fmaf(v.z, v.z, sq); sq = fmaf(v.w, v.w, sq);
        d32[(i >> 1) + 0] = packh2(v.x, v.y);
        d32[(i >> 1) + 1] = packh2(v.z, v.w);
      }
      sq += __shfl_xor(sq, 1, 32);
      if (!(tid & 1)) {
        const size_t gm = (size_t)b * Mc + m;
        float h = logwY[gm] + potScale * potY[gm] - 0.5f * invEps * sq;
        sHH[row] = h * LOG2E;
      }
      if (ch + 1 < nChunks) {  // gfx1250 global_prefetch_b8 of next chunk
        const float* np = Y + (((size_t)b * Mc + mBase + CHUNK_COLS + row) * DIM + c0);
        __builtin_prefetch(np, 0, 1);
      }
    }
    __syncthreads();

    const unsigned* pY = (const unsigned*)sY;
#pragma unroll
    for (int j = 0; j < 8; ++j) {
      HFrag b0, b1;
      const int brow = j * 16 + l15;
#pragma unroll
      for (int v = 0; v < 8; ++v) {
        const int kb = (v < 4 ? 2 * v : 2 * v + 8) + 8 * halfq;
        b0.u[v] = pY[brow * (LDS_STRIDE / 2) + (kb >> 1)];
        b1.u[v] = pY[brow * (LDS_STRIDE / 2) + ((kb + 32) >> 1)];
      }
      v8f c = {};
      c = __builtin_amdgcn_wmma_f32_16x16x32_f16(false, a0.v, false, b0.v,
                                                 (short)0, c, false, false);
      c = __builtin_amdgcn_wmma_f32_16x16x32_f16(false, a1.v, false, b1.v,
                                                 (short)0, c, false, false);
      const float hv = sHH[brow];   // column of this lane's C elements
#pragma unroll
      for (int r = 0; r < 8; ++r) {
        float val = fmaf(c[r], ie2, hv);
        float nm  = fmaxf(runM[r], val);
        runS[r] = fmaf(runS[r], fexp2(runM[r] - nm), fexp2(val - nm));
        runM[r] = nm;
      }
    }
    __syncthreads();
  }

  // ---- reduce the 16 column-slots per half, finish LSE, write potential ----
#pragma unroll
  for (int r = 0; r < 8; ++r) {
#pragma unroll
    for (int off = 1; off < 16; off <<= 1) {
      float oM = __shfl_xor(runM[r], off, 32);
      float oS = __shfl_xor(runS[r], off, 32);
      float nm = fmaxf(runM[r], oM);
      runS[r] = runS[r] * fexp2(runM[r] - nm) + oS * fexp2(oM - nm);
      runM[r] = nm;
    }
  }
  if (l15 == 0) {
    const size_t bo = (size_t)b * Nr;
#pragma unroll
    for (int r = 0; r < 8; ++r) {
      const int rl  = wave * 16 + 8 * halfq + r;
      const int row = rowBase + rl;
      float lse = LN2 * (runM[r] + __log2f(runS[r]));
      float val = 0.5f * sX2[rl] - eps * lse;
      if (mode == 1) val = 0.5f * (prevPot[bo + row] + val);
      outPot[bo + row] = val;
    }
  }
}

// ---- weight normalization: a = w/sum(w), log_a ----------------------------
__global__ void prep_weights(const float* __restrict__ w, float* __restrict__ anorm,
                             float* __restrict__ logw, int n)
{
  __shared__ float red[256];
  const int b = blockIdx.x;
  const float* wb = w + (size_t)b * n;
  float s = 0.f;
  for (int i = threadIdx.x; i < n; i += 256) s += wb[i];
  red[threadIdx.x] = s;
  __syncthreads();
  for (int k = 128; k > 0; k >>= 1) {
    if (threadIdx.x < k) red[threadIdx.x] += red[threadIdx.x + k];
    __syncthreads();
  }
  float mass = red[0];
  if (mass == 0.f) mass = 1.f;
  const float inv = 1.0f / mass;
  for (int i = threadIdx.x; i < n; i += 256) {
    float a = wb[i] * inv;
    anorm[(size_t)b * n + i] = a;
    logw[(size_t)b * n + i] = __logf(a);
  }
}

// ---- final: mean_b [ <a, f_fin - f_aa> + <b, g_fin - g_bb> ] ---------------
__global__ void final_reduce(const float* __restrict__ a, const float* __restrict__ bw,
                             const float* __restrict__ f_fin, const float* __restrict__ f_aa,
                             const float* __restrict__ g_fin, const float* __restrict__ g_bb,
                             float* __restrict__ out, int B, int N, int M)
{
  __shared__ float red[256];
  float s = 0.f;
  for (int i = threadIdx.x; i < B * N; i += 256) s += a[i] * (f_fin[i] - f_aa[i]);
  for (int i = threadIdx.x; i < B * M; i += 256) s += bw[i] * (g_fin[i] - g_bb[i]);
  red[threadIdx.x] = s;
  __syncthreads();
  for (int k = 128; k > 0; k >>= 1) {
    if (threadIdx.x < k) red[threadIdx.x] += red[threadIdx.x + k];
    __syncthreads();
  }
  if (threadIdx.x == 0) out[0] = red[0] / (float)B;
}

extern "C" void kernel_launch(void* const* d_in, const int* in_sizes, int n_in,
                              void* d_out, int out_size, void* d_ws, size_t ws_size,
                              hipStream_t stream)
{
  (void)in_sizes; (void)n_in; (void)out_size; (void)ws_size;
  const float* X  = (const float*)d_in[0];   // [B,N,64]
  const float* Y  = (const float*)d_in[1];   // [B,M,64]
  const float* W1 = (const float*)d_in[2];   // [B,N]
  const float* W2 = (const float*)d_in[3];   // [B,M]
  const int B = 8, N = 2048, M = 2048;

  float* ws = (float*)d_ws;
  const size_t BN = (size_t)B * N;           // == B*M
  float* a_norm = ws + 0 * BN;
  float* b_norm = ws + 1 * BN;
  float* log_a  = ws + 2 * BN;
  float* log_b  = ws + 3 * BN;
  float* fba[2] = { ws + 4 * BN, ws + 5 * BN };
  float* gab[2] = { ws + 6 * BN, ws + 7 * BN };
  float* faa[2] = { ws + 8 * BN, ws + 9 * BN };
  float* gbb[2] = { ws + 10 * BN, ws + 11 * BN };
  float* f_fin = ws + 12 * BN;
  float* g_fin = ws + 13 * BN;
  float* faa_f = ws + 14 * BN;
  float* gbb_f = ws + 15 * BN;

  prep_weights<<<B, 256, 0, stream>>>(W1, a_norm, log_a, N);
  prep_weights<<<B, 256, 0, stream>>>(W2, b_norm, log_b, M);

  // eps schedule: diameter^2 down to blur^2, x(scaling^2)
  double epsl[32]; int ne = 0;
  const double target = 0.05 * 0.05;
  double e = 10.0 * 10.0;
  while (e > target) { epsl[ne++] = e; e *= 0.25; }
  epsl[ne++] = target;            // 9 values

  dim3 grid(N / TILE_ROWS, B), blk(256);
  const float e0 = (float)epsl[0];
  // initialization: h = log weights only (potScale = 0), raw write
  sinkhorn_softmin<<<grid, blk, 0, stream>>>(X, Y, log_b, log_b, 0.f, nullptr, fba[0], e0, N, M, 0);
  sinkhorn_softmin<<<grid, blk, 0, stream>>>(Y, X, log_a, log_a, 0.f, nullptr, gab[0], e0, M, N, 0);
  sinkhorn_softmin<<<grid, blk, 0, stream>>>(X, X, log_a, log_a, 0.f, nullptr, faa[0], e0, N, N, 0);
  sinkhorn_softmin<<<grid, blk, 0, stream>>>(Y, Y, log_b, log_b, 0.f, nullptr, gbb[0], e0, M, M, 0);

  int cur = 0;
  for (int i = 0; i < ne; ++i) {
    const float eps = (float)epsl[i];
    const float ie  = 1.0f / eps;
    const int nxt = cur ^ 1;
    // all four updates read the OLD potentials (ping-pong buffers)
    sinkhorn_softmin<<<grid, blk, 0, stream>>>(X, Y, log_b, gab[cur], ie, fba[cur], fba[nxt], eps, N, M, 1);
    sinkhorn_softmin<<<grid, blk, 0, stream>>>(Y, X, log_a, fba[cur], ie, gab[cur], gab[nxt], eps, M, N, 1);
    sinkhorn_softmin<<<grid, blk, 0, stream>>>(X, X, log_a, faa[cur], ie, faa[cur], faa[nxt], eps, N, N, 1);
    sinkhorn_softmin<<<grid, blk, 0, stream>>>(Y, Y, log_b, gbb[cur], ie, gbb[cur], gbb[nxt], eps, M, M, 1);
    cur = nxt;
  }

  // final non-averaged extrapolation at target eps
  const float ef = (float)epsl[ne - 1];
  const float ie = 1.0f / ef;
  sinkhorn_softmin<<<grid, blk, 0, stream>>>(X, Y, log_b, gab[cur], ie, nullptr, f_fin, ef, N, M, 0);
  sinkhorn_softmin<<<grid, blk, 0, stream>>>(Y, X, log_a, fba[cur], ie, nullptr, g_fin, ef, M, N, 0);
  sinkhorn_softmin<<<grid, blk, 0, stream>>>(X, X, log_a, faa[cur], ie, nullptr, faa_f, ef, N, N, 0);
  sinkhorn_softmin<<<grid, blk, 0, stream>>>(Y, Y, log_b, gbb[cur], ie, nullptr, gbb_f, ef, M, M, 0);

  final_reduce<<<1, 256, 0, stream>>>(a_norm, b_norm, f_fin, faa_f, g_fin, gbb_f,
                                      (float*)d_out, B, N, M);
}
